// CustomTransformerEncoderLayer_79078937854178
// MI455X (gfx1250) — matrix operand order repaired and checked by reading, not compiled, not run on hardware
//
#include <hip/hip_runtime.h>

// ---------------------------------------------------------------------------
// MI455X (gfx1250) Nystrom-attention transformer encoder layer.
// All contractions run on the WMMA pipe:
//   - bf16 16x16x32 WMMA (fp32 accum) for large GEMMs, LDS-staged tiles,
//     128x64 block tile / 32x32 wave tile (4 WMMAs per wave per K-step,
//     2x2 fragment reuse)
//   - fp32 16x16x4 WMMA for the Moore-Penrose Newton-Schulz iterations
// ---------------------------------------------------------------------------

typedef __attribute__((ext_vector_type(16))) __bf16 v16bf;
typedef __attribute__((ext_vector_type(8)))  float  v8f;
typedef __attribute__((ext_vector_type(2)))  float  v2f;

#define HEADS 8
#define DHEAD 64
#define SEQ   4096
#define BATCH 4
#define DIM   512          // HEADS*DHEAD
#define LM    256          // landmarks
#define BH    (BATCH*HEADS)          // 32
#define BHND  ((long)BH*SEQ*DHEAD)   // 8,388,608 floats per (b,h,n,d) tensor

__device__ __forceinline__ unsigned short f2bf(float f) {
  unsigned u = __float_as_uint(f);
  u += 0x7fffu + ((u >> 16) & 1u);   // round-to-nearest-even
  return (unsigned short)(u >> 16);
}

union BfFrag { uint4 q[2]; v16bf v; };

// ---------------------------------------------------------------------------
// Generic bf16 WMMA GEMM:  C = epilogue( A(MxK) * opB(B) )
//   AMODE 0: A row-major [row*lda + k]
//   AMODE 1: A is attn (b,h,n,d) gathered as (b*n, h*d)  (out-proj input)
//   BT 1: B is (N x K) row-major (NT gemm, both K-contiguous)
//   BT 0: B is (K x N) row-major (NN gemm)
//   EPI 0: C = alpha*acc          EPI 1: q/k/v scatter (+0.125 q scale)
//   EPI 2: +bias                  EPI 3: relu(+bias)    EPI 4: +bias+resid
// Block: 256 thr = 8 waves (4 over M x 2 over N); block tile 128x64; KT=32.
// Each wave: 32x32 strip = 2x2 = four 16x16x32 WMMAs per K-step.
// ---------------------------------------------------------------------------
template<int AMODE, int BT, int EPI>
__global__ void __launch_bounds__(256)
gemm_bf16(const float* __restrict__ A, const float* __restrict__ Bm,
          float* __restrict__ C, int K, int lda, int ldb, int ldc,
          long sAb, long sBb, long sCb,
          const float* __restrict__ bias, const float* __restrict__ resid,
          float alpha)
{
  __shared__ __align__(16) unsigned short As[128 * 32];   // 8 KB
  __shared__ __align__(16) unsigned short Bs[64 * 32];    // 4 KB

  const int tid  = threadIdx.x;
  const int wid  = tid >> 5;
  const int lane = tid & 31;
  const int mr   = lane & 15;
  const int half = lane >> 4;

  const long zb = blockIdx.z;
  A  += zb * sAb;  Bm += zb * sBb;  C += zb * sCb;

  const int m0 = blockIdx.y * 128;
  const int n0 = blockIdx.x * 64;
  const int wm = wid & 3;        // wave row   (32 rows each)
  const int wn = wid >> 2;       // wave col   (32 cols each)

  v8f acc[2][2];
#pragma unroll
  for (int i = 0; i < 2; ++i)
#pragma unroll
    for (int j = 0; j < 2; ++j)
      acc[i][j] = v8f{0.f,0.f,0.f,0.f,0.f,0.f,0.f,0.f};

  for (int k0 = 0; k0 < K; k0 += 32) {
    if (AMODE == 0 && (k0 + 32) < K)   // hint next A tile -> global_prefetch_b8
      __builtin_prefetch(&A[(long)(m0 + (tid & 127)) * lda + k0 + 32]);

    // stage 128x32 A-tile (16 elems/thread) and 64x32 B^T-tile (n-major)
    for (int i = tid; i < 128 * 32; i += 256) {
      const int r  = i >> 5;
      const int kk = (i & 31) + k0;
      float a;
      if (AMODE == 0) {
        a = A[(long)(m0 + r) * lda + kk];
      } else {  // gather attn (b,h,n,d) as (b*4096+n , h*64+d)
        const int row = m0 + r;
        a = A[(long)((row >> 12) * HEADS + (kk >> 6)) * (SEQ * DHEAD)
              + (long)(row & (SEQ - 1)) * DHEAD + (kk & 63)];
      }
      As[i] = f2bf(a);
    }
    for (int i = tid; i < 64 * 32; i += 256) {
      const int r  = i >> 5;
      const int kk = (i & 31) + k0;
      float b;
      if (BT) b = Bm[(long)(n0 + r) * ldb + kk];
      else    b = Bm[(long)kk * ldb + (n0 + r)];
      Bs[i] = f2bf(b);
    }
    __syncthreads();

    // A fragments: lane (half h) = row M=mr, K block 16*half .. +15 (contiguous)
    // B fragments: lane = col N=mr, K block 16*half .. +15 (contiguous)
    BfFrag af[2], bf[2];
#pragma unroll
    for (int s = 0; s < 2; ++s) {
      const uint4* arow = (const uint4*)&As[(wm * 32 + s * 16 + mr) * 32];
      af[s].q[0] = arow[2 * half];
      af[s].q[1] = arow[2 * half + 1];
      const uint4* brow = (const uint4*)&Bs[(wn * 32 + s * 16 + mr) * 32];
      bf[s].q[0] = brow[2 * half];
      bf[s].q[1] = brow[2 * half + 1];
    }
#pragma unroll
    for (int ms = 0; ms < 2; ++ms)
#pragma unroll
      for (int ns = 0; ns < 2; ++ns)
        acc[ms][ns] = __builtin_amdgcn_wmma_f32_16x16x32_bf16(
            false, af[ms].v, false, bf[ns].v, (short)0, acc[ms][ns],
            false, false);
    __syncthreads();
  }

  // epilogue: lane col n0+wn*32+ns*16+mr, rows m0+wm*32+ms*16 + e + 8*half
#pragma unroll
  for (int ms = 0; ms < 2; ++ms) {
#pragma unroll
    for (int ns = 0; ns < 2; ++ns) {
      const int cgl = n0 + wn * 32 + ns * 16 + mr;
#pragma unroll
      for (int e = 0; e < 8; ++e) {
        const int rgl = m0 + wm * 32 + ms * 16 + e + 8 * half;
        float vv = acc[ms][ns][e];
        if (EPI == 0) {
          C[(long)rgl * ldc + cgl] = alpha * vv;
        } else if (EPI == 1) {   // scatter into q|k|v (b,h,n,d), scale q
          const int part = cgl >> 9, inner = cgl & 511;
          const int head = inner >> 6, dd = inner & 63;
          const int b = rgl >> 12, n = rgl & (SEQ - 1);
          if (part == 0) vv *= 0.125f;      // 64^-0.5
          C[(long)part * BHND
            + (long)((b * HEADS + head) * SEQ + n) * DHEAD + dd] = vv;
        } else if (EPI == 2) {
          C[(long)rgl * ldc + cgl] = vv + bias[cgl];
        } else if (EPI == 3) {
          const float t = vv + bias[cgl];
          C[(long)rgl * ldc + cgl] = t > 0.f ? t : 0.f;
        } else if (EPI == 4) {
          C[(long)rgl * ldc + cgl] = vv + bias[cgl] + resid[(long)rgl * ldc + cgl];
        }
      }
    }
  }
}

// ---------------------------------------------------------------------------
// fp32 WMMA GEMM for pinv: 256x256x256 batched (32), NN layout.
//   out = form ? (cdiag*I - A@B) : alpha*(A@B)
// Block 256 thr = 8 waves over a 64x32 region; grid (8,4,32).
// ---------------------------------------------------------------------------
__global__ void __launch_bounds__(256)
pinv_gemm(const float* __restrict__ A, const float* __restrict__ B,
          float* __restrict__ C, int form, float cdiag, float alpha)
{
  const int tid = threadIdx.x, wid = tid >> 5, lane = tid & 31;
  const int mr = lane & 15, half = lane >> 4;
  const long bo = (long)blockIdx.z * (LM * LM);
  A += bo; B += bo; C += bo;
  const int tm = blockIdx.y * 64 + (wid & 3) * 16;
  const int tn = blockIdx.x * 32 + (wid >> 2) * 16;

  v8f acc = {0.f,0.f,0.f,0.f,0.f,0.f,0.f,0.f};
  const float* arow = A + (long)(tm + mr) * LM;
  for (int k = 0; k < LM; k += 4) {
    // A 16x4: lane = row mr, K = k + 2*half + {0,1}  (contiguous pair)
    v2f a = *(const v2f*)&arow[k + 2 * half];
    // B 4x16: lane = col mr, K = k + 2*half + {0,1}  (stride-256 pair)
    v2f b;
    b.x = B[(long)(k + 2 * half)     * LM + tn + mr];
    b.y = B[(long)(k + 2 * half + 1) * LM + tn + mr];
    acc = __builtin_amdgcn_wmma_f32_16x16x4_f32(false, a, false, b,
                                                (short)0, acc, false, false);
  }
  const int col = tn + mr;
#pragma unroll
  for (int e = 0; e < 8; ++e) {
    const int row = tm + e + 8 * half;
    float v = acc[e];
    v = form ? ((row == col ? cdiag : 0.f) - v) : alpha * v;
    C[(long)row * LM + col] = v;
  }
}

// ---------------------------------------------------------------------------
// small fp32 helper kernels
// ---------------------------------------------------------------------------
__global__ void __launch_bounds__(256)
softmax_rows(float* __restrict__ x, int W)
{
  __shared__ float red[256];
  const int tid = threadIdx.x;
  float* row = x + (long)blockIdx.x * W;
  float m = -3.4e38f;
  for (int i = tid; i < W; i += 256) m = fmaxf(m, row[i]);
  red[tid] = m; __syncthreads();
  for (int s = 128; s; s >>= 1) { if (tid < s) red[tid] = fmaxf(red[tid], red[tid + s]); __syncthreads(); }
  m = red[0]; __syncthreads();
  float sum = 0.f;
  for (int i = tid; i < W; i += 256) { float e = __expf(row[i] - m); row[i] = e; sum += e; }
  red[tid] = sum; __syncthreads();
  for (int s = 128; s; s >>= 1) { if (tid < s) red[tid] += red[tid + s]; __syncthreads(); }
  const float inv = 1.f / red[0];
  for (int i = tid; i < W; i += 256) row[i] *= inv;
}

__global__ void __launch_bounds__(256)
layernorm_rows(const float* __restrict__ x, const float* __restrict__ g,
               const float* __restrict__ b, float* __restrict__ out)
{
  __shared__ float red[256];
  const int tid = threadIdx.x;
  const long base = (long)blockIdx.x * DIM;
  float s = 0.f;
  for (int i = tid; i < DIM; i += 256) s += x[base + i];
  red[tid] = s; __syncthreads();
  for (int t = 128; t; t >>= 1) { if (tid < t) red[tid] += red[tid + t]; __syncthreads(); }
  const float mu = red[0] * (1.f / DIM); __syncthreads();
  float s2 = 0.f;
  for (int i = tid; i < DIM; i += 256) { float d = x[base + i] - mu; s2 += d * d; }
  red[tid] = s2; __syncthreads();
  for (int t = 128; t; t >>= 1) { if (tid < t) red[tid] += red[tid + t]; __syncthreads(); }
  const float inv = rsqrtf(red[0] * (1.f / DIM) + 1e-5f);
  for (int i = tid; i < DIM; i += 256)
    out[base + i] = (x[base + i] - mu) * inv * g[i] + b[i];
}

__global__ void __launch_bounds__(64)
landmark_mean(const float* __restrict__ x, float* __restrict__ out)
{
  const int bh = blockIdx.y, m = blockIdx.x, dd = threadIdx.x;
  const float* p = x + ((long)bh * SEQ + m * 16) * DHEAD + dd;
  float s = 0.f;
#pragma unroll
  for (int i = 0; i < 16; ++i) s += p[i * DHEAD];
  out[((long)bh * LM + m) * DHEAD + dd] = s * (1.f / 16.f);
}

__global__ void init_scalars(float* s) { if (threadIdx.x < 2) s[threadIdx.x] = 0.f; }

__global__ void __launch_bounds__(64)
abs_sum_max(const float* __restrict__ a2, float* __restrict__ scal, int axis)
{
  __shared__ float red[64];
  const int bh = blockIdx.y, i = blockIdx.x, tid = threadIdx.x;
  const float* mat = a2 + (long)bh * (LM * LM);
  float s = 0.f;
  for (int j = tid; j < LM; j += 64)
    s += fabsf(axis == 0 ? mat[i * LM + j] : mat[j * LM + i]);
  red[tid] = s; __syncthreads();
  for (int t = 32; t; t >>= 1) { if (tid < t) red[tid] += red[tid + t]; __syncthreads(); }
  if (tid == 0) atomicMax((int*)&scal[axis], __float_as_int(red[0]));  // positive floats
}

__global__ void __launch_bounds__(256)
transpose_scale(const float* __restrict__ a2, const float* __restrict__ scal,
                float* __restrict__ z)
{
  const float s = 1.f / (scal[0] * scal[1]);
  const int bh = blockIdx.y, i = blockIdx.x, j = threadIdx.x;
  z[(long)bh * (LM * LM) + i * LM + j] = a2[(long)bh * (LM * LM) + j * LM + i] * s;
}

__global__ void __launch_bounds__(256)
identity_minus(const float* __restrict__ x, float* __restrict__ t, float c)
{
  const long idx = (long)blockIdx.x * 256 + threadIdx.x;   // 32*65536 total
  const int ij = (int)(idx & (LM * LM - 1));
  const int i = ij >> 8, j = ij & 255;
  t[idx] = (i == j ? c : 0.f) - x[idx];
}

__global__ void __launch_bounds__(64)
dwconv_add(const float* __restrict__ v, const float* __restrict__ wgt,
           float* __restrict__ attn)
{
  const int n = blockIdx.x, bh = blockIdx.y, dd = threadIdx.x;
  const int h = bh & (HEADS - 1);
  const float* vb = v + (long)bh * SEQ * DHEAD;
  float s = 0.f;
#pragma unroll
  for (int t = 0; t < 33; ++t) {
    const int nn = n - 16 + t;
    if (nn >= 0 && nn < SEQ) s += vb[(long)nn * DHEAD + dd] * wgt[h * 33 + t];
  }
  attn[(long)bh * SEQ * DHEAD + (long)n * DHEAD + dd] += s;
}

// ---------------------------------------------------------------------------
extern "C" void kernel_launch(void* const* d_in, const int* in_sizes, int n_in,
                              void* d_out, int out_size, void* d_ws, size_t ws_size,
                              hipStream_t stream)
{
  (void)in_sizes; (void)n_in; (void)out_size; (void)ws_size;
  const float* src   = (const float*)d_in[0];
  const float* w_qkv = (const float*)d_in[1];
  const float* w_out = (const float*)d_in[2];
  const float* b_out = (const float*)d_in[3];
  const float* convw = (const float*)d_in[4];
  const float* w1    = (const float*)d_in[5];
  const float* b1    = (const float*)d_in[6];
  const float* w2    = (const float*)d_in[7];
  const float* b2    = (const float*)d_in[8];
  const float* g1    = (const float*)d_in[9];
  const float* be1   = (const float*)d_in[10];
  const float* g2    = (const float*)d_in[11];
  const float* be2   = (const float*)d_in[12];

  float* w = (float*)d_ws;
  size_t off = 0;
  auto alloc = [&](size_t n) { float* p = w + off; off += n; return p; };

  float* q    = alloc(BHND);                 // qkv gemm writes q|k|v contiguously
  float* k    = alloc(BHND);
  float* v    = alloc(BHND);
  float* ql   = alloc((size_t)BH * LM * DHEAD);
  float* kl   = alloc((size_t)BH * LM * DHEAD);
  float* a1   = alloc((size_t)BH * SEQ * LM);   // later reused for x1/xn1/h1/x2
  float* a3   = alloc((size_t)BH * LM * SEQ);   // later reused for az
  float* a2   = alloc((size_t)BH * LM * LM);
  float* zbuf = alloc((size_t)BH * LM * LM);
  float* znb  = alloc((size_t)BH * LM * LM);
  float* xz   = alloc((size_t)BH * LM * LM);
  float* t1   = alloc((size_t)BH * LM * LM);
  float* t2   = alloc((size_t)BH * LM * LM);
  float* a3v  = alloc((size_t)BH * LM * DHEAD);
  float* attn = alloc(BHND);
  float* scal = alloc(16);
  (void)k;

  const dim3 blk(256);
  const long sQ  = (long)SEQ * DHEAD;   // 262144
  const long sL  = (long)LM * DHEAD;    // 16384
  const long sA1 = (long)SEQ * LM;      // 1048576
  const long sA2 = (long)LM * LM;       // 65536

  // 1. QKV projection, fused head scatter + q scale (NT, K=512)
  gemm_bf16<0,1,1><<<dim3(1536/64, 16384/128, 1), blk, 0, stream>>>(
      src, w_qkv, q, 512, 512, 512, 0, 0L, 0L, 0L, nullptr, nullptr, 1.f);

  // 2. landmark means
  landmark_mean<<<dim3(LM, BH), 64, 0, stream>>>(q, ql);
  landmark_mean<<<dim3(LM, BH), 64, 0, stream>>>(k, kl);

  // 3. similarity GEMMs (NT, K=64, batched over 32 (b,h))
  gemm_bf16<0,1,0><<<dim3(LM/64, SEQ/128, BH), blk, 0, stream>>>(
      q,  kl, a1, 64, 64, 64, LM,  sQ, sL, sA1, nullptr, nullptr, 1.f);
  gemm_bf16<0,1,0><<<dim3(LM/64, LM/128, BH), blk, 0, stream>>>(
      ql, kl, a2, 64, 64, 64, LM,  sL, sL, sA2, nullptr, nullptr, 1.f);
  gemm_bf16<0,1,0><<<dim3(SEQ/64, LM/128, BH), blk, 0, stream>>>(
      ql, k,  a3, 64, 64, 64, SEQ, sL, sQ, sA1, nullptr, nullptr, 1.f);

  // 4. softmaxes
  softmax_rows<<<dim3(BH * SEQ), blk, 0, stream>>>(a1, LM);
  softmax_rows<<<dim3(BH * LM),  blk, 0, stream>>>(a2, LM);
  softmax_rows<<<dim3(BH * LM),  blk, 0, stream>>>(a3, SEQ);

  // 5. pinv init: global max row/col abs sums, z = a2^T / (maxc*maxr)
  init_scalars<<<1, 32, 0, stream>>>(scal);
  abs_sum_max<<<dim3(LM, BH), 64, 0, stream>>>(a2, scal, 0);
  abs_sum_max<<<dim3(LM, BH), 64, 0, stream>>>(a2, scal, 1);
  transpose_scale<<<dim3(LM, BH), blk, 0, stream>>>(a2, scal, zbuf);

  // 6. Newton-Schulz pinv, fp32 WMMA (4 matmuls/iter, 256^3 x 32 batch)
  float* zc = zbuf; float* zn = znb;
  const dim3 pg(LM/32, LM/64, BH);
  for (int it = 0; it < 6; ++it) {
    pinv_gemm<<<pg, blk, 0, stream>>>(a2, zc, xz, 0, 0.f, 1.f);       // xz = a2@z
    identity_minus<<<dim3(BH * sA2 / 256), blk, 0, stream>>>(xz, t1, 7.f);
    pinv_gemm<<<pg, blk, 0, stream>>>(xz, t1, t2, 1, 15.f, 1.f);      // 15I - xz@t1
    pinv_gemm<<<pg, blk, 0, stream>>>(xz, t2, t1, 1, 13.f, 1.f);      // 13I - xz@t2
    pinv_gemm<<<pg, blk, 0, stream>>>(zc, t1, zn, 0, 0.f, 0.25f);     // z = .25 z@t1
    float* tmp = zc; zc = zn; zn = tmp;
  }

  // 7. a3v = a3 @ v   (NN, K=4096)
  gemm_bf16<0,0,0><<<dim3(DHEAD/64, LM/128, BH), blk, 0, stream>>>(
      a3, v, a3v, SEQ, SEQ, DHEAD, DHEAD, sA1, sQ, sL, nullptr, nullptr, 1.f);

  // 8. az = a1 @ a2inv  (NN, K=256)  -- az aliases a3 (a3 now dead)
  float* az = a3;
  gemm_bf16<0,0,0><<<dim3(LM/64, SEQ/128, BH), blk, 0, stream>>>(
      a1, zc, az, LM, LM, LM, LM, sA1, sA2, sA1, nullptr, nullptr, 1.f);

  // 9. attn = az @ a3v  (NN, K=256)
  gemm_bf16<0,0,0><<<dim3(DHEAD/64, SEQ/128, BH), blk, 0, stream>>>(
      az, a3v, attn, LM, LM, DHEAD, DHEAD, sA1, sL, sQ, nullptr, nullptr, 1.f);

  // 10. depthwise conv-33 residual on v, added into attn
  dwconv_add<<<dim3(SEQ, BH), 64, 0, stream>>>(v, convw, attn);

  // 11. out-proj with fused bias + src residual (A gathered from (b,h,n,d))
  float* x1  = a1;                    // a1 region reused: 4 x 8.39M floats
  float* xn1 = a1 + BHND;
  float* h1  = a1 + 2 * BHND;
  float* x2  = a1 + 3 * BHND;
  gemm_bf16<1,1,4><<<dim3(DIM/64, 16384/128, 1), blk, 0, stream>>>(
      attn, w_out, x1, 512, 0, 512, 512, 0L, 0L, 0L, b_out, src, 1.f);

  // 12. LN1
  layernorm_rows<<<dim3(BATCH * SEQ), blk, 0, stream>>>(x1, g1, be1, xn1);

  // 13. FFN: relu(xn1@w1^T + b1) @ w2^T + b2 + xn1
  gemm_bf16<0,1,3><<<dim3(DIM/64, 16384/128, 1), blk, 0, stream>>>(
      xn1, w1, h1, 512, 512, 512, 512, 0L, 0L, 0L, b1, nullptr, 1.f);
  gemm_bf16<0,1,4><<<dim3(DIM/64, 16384/128, 1), blk, 0, stream>>>(
      h1, w2, x2, 512, 512, 512, 512, 0L, 0L, 0L, b2, xn1, 1.f);

  // 14. LN2 -> output
  layernorm_rows<<<dim3(BATCH * SEQ), blk, 0, stream>>>(x2, g2, be2, (float*)d_out);
}
